// ShapeDynamicFeedForward_36266703847579
// MI455X (gfx1250) — matrix operand order repaired and checked by reading, not compiled, not run on hardware
//
#include <hip/hip_runtime.h>
#include <hip/hip_bf16.h>

typedef __bf16 bf16_t;
typedef __attribute__((ext_vector_type(16))) __bf16 v16bf;
typedef __attribute__((ext_vector_type(8)))  __bf16 v8bf;
typedef __attribute__((ext_vector_type(8)))  float  v8f;

// problem dims
#define N_    8
#define CIN_  64
#define DIMS_ 64
#define NW_   8
#define T_    16
#define H_    64
#define W_    64
#define CNT_PER_CH 524288.0f   // N*T*H*W
#define BN_EPS 1e-5f

// padded channel-last activation tensor: [N][T+2][H+2][W+2][C]
#define TP_ 18
#define HP_ 66
#define WP_ 66
#define XPAD_ELEMS (N_ * TP_ * HP_ * WP_ * CIN_)   // 40,144,896 bf16 = 80,289,792 B

// ---------------------------------------------------------------------------
// K0a: zero padded activation buffer (halo must be 0; interior overwritten)
// ---------------------------------------------------------------------------
__global__ __launch_bounds__(256) void k_zero_pad(uint4* __restrict__ p, int n16) {
    int i = blockIdx.x * 256 + threadIdx.x;
    if (i < n16) p[i] = uint4{0u, 0u, 0u, 0u};
}

// ---------------------------------------------------------------------------
// K0b: zero the scratch accumulators (pooled max, BN sum/sumsq)
// ---------------------------------------------------------------------------
__global__ void k_init(float* __restrict__ pooled, float* __restrict__ chsum,
                       float* __restrict__ chsq) {
    int t = threadIdx.x;
    if (t < 256) pooled[t] = 0.0f;        // 8*32
    if (t < 64)  { chsum[t] = 0.0f; chsq[t] = 0.0f; }
}

// ---------------------------------------------------------------------------
// K1: std over temporal axis (ddof=1): x[n,c,t,h,w] -> stdx[n,c,h,w]
// ---------------------------------------------------------------------------
__global__ __launch_bounds__(256) void k_std(const float* __restrict__ x,
                                             float* __restrict__ stdx) {
    int idx = blockIdx.x * 256 + threadIdx.x;          // over N*C*H*W = 2M
    if (idx >= N_ * CIN_ * H_ * W_) return;
    int w = idx & 63, h = (idx >> 6) & 63, c = (idx >> 12) & 63, n = idx >> 18;
    const float* p = x + ((((n * CIN_ + c) * T_) * H_ + h) * W_) + w;
    float vals[T_];
    float s = 0.0f;
#pragma unroll
    for (int t = 0; t < T_; ++t) { vals[t] = p[t * (H_ * W_)]; s += vals[t]; }
    float mean = s * (1.0f / T_);
    float ss = 0.0f;
#pragma unroll
    for (int t = 0; t < T_; ++t) { float d = vals[t] - mean; ss += d * d; }
    stdx[((n * CIN_ + c) * H_ + h) * W_ + w] = sqrtf(ss * (1.0f / (T_ - 1)));
}

// ---------------------------------------------------------------------------
// K2: LDS-tiled transpose to channel-last bf16 into the PADDED buffer:
//     xclp[n][t+1][h+1][w+1][c]
// ---------------------------------------------------------------------------
__global__ __launch_bounds__(256) void k_transpose(const float* __restrict__ x,
                                                   bf16_t* __restrict__ xclp) {
    __shared__ float tile[64][33];
    int bid = blockIdx.x;                  // n*t*h*2 = 16384 blocks
    int wseg = bid & 1;
    int h = (bid >> 1) & 63;
    int t = (bid >> 7) & 15;
    int n = bid >> 11;
    int w0 = wseg * 32;

    int wCol = threadIdx.x & 31;
    int cRow = threadIdx.x >> 5;           // 0..7
#pragma unroll
    for (int cc = cRow; cc < 64; cc += 8)
        tile[cc][wCol] = x[((((n * CIN_ + cc) * T_ + t) * H_ + h) * W_) + w0 + wCol];
    __syncthreads();

    int cCol = threadIdx.x & 63;
    int wRow = threadIdx.x >> 6;           // 0..3
    bf16_t* orow = xclp + (((n * TP_ + (t + 1)) * HP_ + (h + 1)) * WP_) * CIN_;
#pragma unroll
    for (int ww = wRow; ww < 32; ww += 4)
        orow[(w0 + ww + 1) * CIN_ + cCol] = (bf16_t)tile[cCol][ww];
}

// ---------------------------------------------------------------------------
// K3: SE conv1 5x5 VALID, 64->16 ch, + bias + relu : h1[n,16,60,60]
// ---------------------------------------------------------------------------
__global__ __launch_bounds__(256) void k_se1(const float* __restrict__ stdx,
                                             const float* __restrict__ w1,
                                             const float* __restrict__ b1,
                                             float* __restrict__ h1) {
    int idx = blockIdx.x * 256 + threadIdx.x;
    if (idx >= 8 * 16 * 60 * 60) return;
    int ow = idx % 60;
    int oh = (idx / 60) % 60;
    int oc = (idx / 3600) % 16;
    int n  = idx / 57600;
    float acc = b1[oc];
    for (int c = 0; c < CIN_; ++c) {
        const float* xp = stdx + ((n * CIN_ + c) * 64 + oh) * 64 + ow;
        const float* wp = w1 + (oc * CIN_ + c) * 25;
#pragma unroll
        for (int ky = 0; ky < 5; ++ky)
#pragma unroll
            for (int kx = 0; kx < 5; ++kx)
                acc += xp[ky * 64 + kx] * wp[ky * 5 + kx];
    }
    h1[idx] = fmaxf(acc, 0.0f);
}

// ---------------------------------------------------------------------------
// K4: SE conv2 5x5 VALID 16->32 + relu, fused global spatial max-pool
// ---------------------------------------------------------------------------
__global__ __launch_bounds__(256) void k_se2_pool(const float* __restrict__ h1,
                                                  const float* __restrict__ w2,
                                                  const float* __restrict__ b2,
                                                  float* __restrict__ pooled) {
    int idx = blockIdx.x * 256 + threadIdx.x;
    if (idx >= 8 * 32 * 56 * 56) return;
    int ow = idx % 56;
    int oh = (idx / 56) % 56;
    int oc = (idx / 3136) % 32;
    int n  = idx / 100352;
    float acc = b2[oc];
    for (int c = 0; c < 16; ++c) {
        const float* xp = h1 + ((n * 16 + c) * 60 + oh) * 60 + ow;
        const float* wp = w2 + (oc * 16 + c) * 25;
#pragma unroll
        for (int ky = 0; ky < 5; ++ky)
#pragma unroll
            for (int kx = 0; kx < 5; ++kx)
                acc += xp[ky * 60 + kx] * wp[ky * 5 + kx];
    }
    acc = fmaxf(acc, 0.0f);
    // non-negative floats: uint bit pattern is order-preserving
    atomicMax((unsigned int*)&pooled[n * 32 + oc], __float_as_uint(acc));
}

// ---------------------------------------------------------------------------
// K5: head: logits -> softmax(tau) -> phi ; dyn_b = phi @ biases
// ---------------------------------------------------------------------------
__global__ __launch_bounds__(64) void k_head(const float* __restrict__ pooled,
                                             const float* __restrict__ lin_w,
                                             const float* __restrict__ lin_b,
                                             const float* __restrict__ biases,
                                             const int* __restrict__ ep,
                                             float* __restrict__ phi_g,
                                             float* __restrict__ dynb) {
    __shared__ float logits[64];
    __shared__ float phiL[64];
    int tid = threadIdx.x;
    int n = tid >> 3, j = tid & 7;
    float acc = lin_b[j];
#pragma unroll
    for (int k = 0; k < 32; ++k)
        acc += pooled[n * 32 + k] * lin_w[j * 32 + k];
    logits[tid] = acc;
    __syncthreads();
    if (tid < 8) {
        int e = ep[0];
        float tau = (e < 10) ? (30.0f - 2.9f * (float)e) : 1.0f;
        float mx = logits[tid * 8];
#pragma unroll
        for (int jj = 1; jj < 8; ++jj) mx = fmaxf(mx, logits[tid * 8 + jj]);
        float v[8], s = 0.0f;
#pragma unroll
        for (int jj = 0; jj < 8; ++jj) {
            v[jj] = expf((logits[tid * 8 + jj] - mx) / tau);
            s += v[jj];
        }
        float inv = 1.0f / s;
#pragma unroll
        for (int jj = 0; jj < 8; ++jj) {
            phiL[tid * 8 + jj] = v[jj] * inv;
            phi_g[tid * 8 + jj] = v[jj] * inv;
        }
    }
    __syncthreads();
    // dyn_b[n][d] = sum_k phi[n][k] * biases[k][d]  (tid = d)
    for (int nn = 0; nn < 8; ++nn) {
        float a2 = 0.0f;
#pragma unroll
        for (int k = 0; k < 8; ++k)
            a2 += phiL[nn * 8 + k] * biases[k * DIMS_ + tid];
        dynb[nn * DIMS_ + tid] = a2;
    }
}

// ---------------------------------------------------------------------------
// K6: dyn_w mix -> bf16 in FRAGMENT-LINEAR layout:
//   dynw[(((n*27+tap)*2+chunk)*4+mt)*512 + lane*16 + e]
//   maps to weight element (m, c):
//     m = mt*16 + (lane&15)
//     c = chunk*32 + (lane>>4)*16 + e   (contiguous K per half, matches B)
// weights layout: [nw][oc][c][3][3][3] (tap = dz*9+dy*3+dx contiguous)
// ---------------------------------------------------------------------------
__global__ __launch_bounds__(256) void k_dynw(const float* __restrict__ phi,
                                              const float* __restrict__ weights,
                                              bf16_t* __restrict__ dynw) {
    int idx = blockIdx.x * 256 + threadIdx.x;      // < 8*27*2*4*32*16 = 884736
    if (idx >= N_ * 27 * DIMS_ * CIN_) return;
    int e     = idx & 15;
    int lane  = (idx >> 4) & 31;
    int mt    = (idx >> 9) & 3;
    int chunk = (idx >> 11) & 1;
    int rest  = idx >> 12;                         // n*27 + tap
    int tap   = rest % 27;
    int n     = rest / 27;
    int m = mt * 16 + (lane & 15);
    int c = chunk * 32 + ((lane >> 4) << 4) + e;
    float acc = 0.0f;
#pragma unroll
    for (int k = 0; k < NW_; ++k)
        acc += phi[n * NW_ + k] * weights[(((k * DIMS_ + m) * CIN_ + c) * 27) + tap];
    dynw[idx] = (bf16_t)acc;
}

// ---------------------------------------------------------------------------
// K7: main per-sample conv3d (SAME) as implicit GEMM on bf16 WMMA.
//   1 wave per block; grid = N*T*H = 8192 blocks
//   wave -> M=64 (4 M-tiles) x N=64 w-positions (4 N-tiles, full row)
//   padded (halo-zeroed) input => ALL loads unconditional, no EXEC juggling
//   per tap: 2 chunks x (4 A-loads + 4 B-loads + 16 WMMA)
// ---------------------------------------------------------------------------
__global__ __launch_bounds__(32) void k_conv3d_wmma(
    const bf16_t* __restrict__ xclp, const bf16_t* __restrict__ dynw,
    const float* __restrict__ dynb, float* __restrict__ yout,
    float* __restrict__ chsum, float* __restrict__ chsq) {
    const int bid = blockIdx.x;
    const int h = bid & 63;
    const int t = (bid >> 6) & 15;
    const int n = bid >> 10;
    const int lane = threadIdx.x & 31;
    const int nIdx = lane & 15;
    const int hf   = lane >> 4;          // which half-wave
    const int kbB  = hf * 16;            // B frag K-base (16 contiguous K per half)

    v8f acc[4][4];
#pragma unroll
    for (int mt = 0; mt < 4; ++mt)
#pragma unroll
        for (int nt = 0; nt < 4; ++nt) acc[mt][nt] = {};

    // fragment-linear A: per (n,tap,chunk,mt): 512 elems; this lane's 16 at +lane*16
    const bf16_t* abase = dynw + n * (27 * 2 * 4 * 512) + lane * 16;
    const bf16_t* xbase = xclp + n * (TP_ * HP_ * WP_ * CIN_);

    int tap = 0;
    for (int dz = 0; dz < 3; ++dz) {
        const bf16_t* xt = xbase + (t + dz) * (HP_ * WP_ * CIN_);
        for (int dy = 0; dy < 3; ++dy) {
            const bf16_t* xrow = xt + (h + dy) * (WP_ * CIN_);
            for (int dx = 0; dx < 3; ++dx, ++tap) {
                const bf16_t* afrag = abase + tap * (2 * 4 * 512);
                const bf16_t* brow = xrow + (nIdx + dx) * CIN_ + kbB;
#pragma unroll
                for (int chunk = 0; chunk < 2; ++chunk) {
                    v16bf a[4];
#pragma unroll
                    for (int mt = 0; mt < 4; ++mt)
                        a[mt] = *(const v16bf*)(afrag + (chunk * 4 + mt) * 512);
                    v16bf b[4];
#pragma unroll
                    for (int nt = 0; nt < 4; ++nt)
                        b[nt] = *(const v16bf*)(brow + chunk * 32 + nt * (16 * CIN_));
#pragma unroll
                    for (int mt = 0; mt < 4; ++mt)
#pragma unroll
                        for (int nt = 0; nt < 4; ++nt)
                            acc[mt][nt] = __builtin_amdgcn_wmma_f32_16x16x32_bf16(
                                false, a[mt], false, b[nt], (short)0, acc[mt][nt],
                                false, false);
                }
            }
        }
    }

    // epilogue: bias add, store pre-BN y, per-channel sum/sumsq reduction
#pragma unroll
    for (int mt = 0; mt < 4; ++mt) {
#pragma unroll
        for (int v = 0; v < 8; ++v) {
            const int m = mt * 16 + hf * 8 + v;   // C/D layout: m = v + 8*(lane>=16)
            const float bias = dynb[n * DIMS_ + m];
            float* orow = yout + ((((n * DIMS_ + m) * T_ + t) * H_ + h) * W_);
            float s = 0.0f, sq = 0.0f;
#pragma unroll
            for (int nt = 0; nt < 4; ++nt) {
                const int w = nt * 16 + nIdx;
                const float val = acc[mt][nt][v] + bias;
                orow[w] = val;
                s += val;
                sq += val * val;
            }
#pragma unroll
            for (int d = 1; d < 16; d <<= 1) {
                s  += __shfl_xor(s, d, 32);
                sq += __shfl_xor(sq, d, 32);
            }
            if (nIdx == 0) {
                atomicAdd(&chsum[m], s);
                atomicAdd(&chsq[m], sq);
            }
        }
    }
}

// ---------------------------------------------------------------------------
// K8: BatchNorm (batch stats, biased var) + ReLU, in place, float4 vectorized
// ---------------------------------------------------------------------------
__global__ __launch_bounds__(256) void k_bn_relu(float* __restrict__ y,
                                                 const float* __restrict__ chsum,
                                                 const float* __restrict__ chsq,
                                                 const float* __restrict__ gamma,
                                                 const float* __restrict__ beta) {
    int i4 = blockIdx.x * 256 + threadIdx.x;   // float4 index; T*H*W = 65536
    int m = (i4 >> 14) & 63;
    const float inv = 1.0f / CNT_PER_CH;
    float mean = chsum[m] * inv;
    float var  = chsq[m] * inv - mean * mean;
    float sc = gamma[m] * rsqrtf(var + BN_EPS);
    float sh = beta[m] - mean * sc;
    float4* p = (float4*)y + i4;
    float4 v = *p;
    v.x = fmaxf(v.x * sc + sh, 0.0f);
    v.y = fmaxf(v.y * sc + sh, 0.0f);
    v.z = fmaxf(v.z * sc + sh, 0.0f);
    v.w = fmaxf(v.w * sc + sh, 0.0f);
    *p = v;
}

// ---------------------------------------------------------------------------
extern "C" void kernel_launch(void* const* d_in, const int* in_sizes, int n_in,
                              void* d_out, int out_size, void* d_ws, size_t ws_size,
                              hipStream_t stream) {
    const float* x       = (const float*)d_in[0];
    const float* weights = (const float*)d_in[1];
    const float* biases  = (const float*)d_in[2];
    const float* se_w1   = (const float*)d_in[3];
    const float* se_b1   = (const float*)d_in[4];
    const float* se_w2   = (const float*)d_in[5];
    const float* se_b2   = (const float*)d_in[6];
    const float* lin_w   = (const float*)d_in[7];
    const float* lin_b   = (const float*)d_in[8];
    const float* gamma   = (const float*)d_in[9];
    const float* beta    = (const float*)d_in[10];
    const int*   ep      = (const int*)d_in[11];
    float* out = (float*)d_out;

    char* ws = (char*)d_ws;
    bf16_t* xclp  = (bf16_t*)(ws);                 // 80,289,792 B (padded, halo=0)
    float*  stdx  = (float*)(ws + 80289792);       //  8,388,608 B
    float*  h1    = (float*)(ws + 88678400);       //  1,843,200 B
    bf16_t* dynw  = (bf16_t*)(ws + 90521600);      //  1,769,472 B
    float*  pooled= (float*)(ws + 92291072);       //      1,024 B
    float*  phi   = (float*)(ws + 92292096);       //        256 B
    float*  dynb  = (float*)(ws + 92292352);       //      2,048 B
    float*  chsum = (float*)(ws + 92294400);       //        256 B
    float*  chsq  = (float*)(ws + 92294656);       //        256 B

    k_zero_pad<<<19602, 256, 0, stream>>>((uint4*)xclp, XPAD_ELEMS / 8);
    k_init<<<1, 256, 0, stream>>>(pooled, chsum, chsq);
    k_std<<<8192, 256, 0, stream>>>(x, stdx);
    k_transpose<<<16384, 256, 0, stream>>>(x, xclp);
    k_se1<<<1800, 256, 0, stream>>>(stdx, se_w1, se_b1, h1);
    k_se2_pool<<<3136, 256, 0, stream>>>(h1, se_w2, se_b2, pooled);
    k_head<<<1, 64, 0, stream>>>(pooled, lin_w, lin_b, biases, ep, phi, dynb);
    k_dynw<<<3456, 256, 0, stream>>>(phi, weights, dynw);
    k_conv3d_wmma<<<8192, 32, 0, stream>>>(xclp, dynw, dynb, out, chsum, chsq);
    k_bn_relu<<<32768, 256, 0, stream>>>(out, chsum, chsq, gamma, beta);
}
